// GCN_70729521430717
// MI455X (gfx1250) — compile-verified, matrix operand order checked by reading, not run on hardware
//
#include <hip/hip_runtime.h>
#include <hip/hip_bf16.h>
#include <stdint.h>

typedef __attribute__((ext_vector_type(4)))  _Float16 v4h;
typedef __attribute__((ext_vector_type(8)))  _Float16 v8h;
typedef __attribute__((ext_vector_type(16))) _Float16 v16h;
typedef __attribute__((ext_vector_type(8)))  float    v8f;

#define GCN_N_NODES  50000
#define GCN_N_EDGES  800000
#define GCN_DIM      128
#define GCN_LAYERS   3
#define GCN_GRAPHS   64

// ---------------- utility kernels ----------------

__global__ void gcn_fill_f32(float* __restrict__ p, float v, int n) {
    int i = blockIdx.x * blockDim.x + threadIdx.x;
    if (i < n) p[i] = v;
}

// xnext[n][d] = bias[d]  (so the edge scatter can be pure atomicAdd)
__global__ void gcn_fill_bias(float* __restrict__ xn, const float* __restrict__ b, int n) {
    int i = blockIdx.x * blockDim.x + threadIdx.x;
    if (i < n) xn[i] = b[i & (GCN_DIM - 1)];
}

// deg accumulation over edge targets (self loops folded in later as +1)
__global__ void gcn_deg_accum(const long long* __restrict__ col, float* __restrict__ deg, int e) {
    int i = blockIdx.x * blockDim.x + threadIdx.x;
    if (i < e) atomicAdd(deg + (int)col[i], 1.0f);
}

// in-place: deg -> rsqrt(deg + 1)   (self-loop makes deg >= 1 always)
__global__ void gcn_make_dinv(float* __restrict__ d, int n) {
    int i = blockIdx.x * blockDim.x + threadIdx.x;
    if (i < n) d[i] = rsqrtf(d[i] + 1.0f);
}

// f32 -> f16, 4 elements per thread (16B in, 8B out)
__global__ void gcn_cvt_f16(const float* __restrict__ in, _Float16* __restrict__ out, int n4) {
    int i = blockIdx.x * blockDim.x + threadIdx.x;
    if (i < n4) {
        float4 v = ((const float4*)in)[i];
        v4h o = { (_Float16)v.x, (_Float16)v.y, (_Float16)v.z, (_Float16)v.w };
        ((v4h*)out)[i] = o;
    }
}

// ---------------- WMMA GEMM: h[N,128] = xh[N,128] @ wh[128,128]^T ----------------
// h[m,n] = sum_k xh[m,k] * wh[n,k]
// block = 256 threads = 8 waves; wave w computes the 16x16 tile at (m0 = blockIdx*16, n0 = 16*w)
// A fragment (16-bit, 16x32): lanes 0-15 hold row m=lane, halves -> K {0..7,16..23};
// lanes 16-31 hold the same rows, halves -> K {8..15,24..31}.  B mirrors with lane = N column.
__global__ __launch_bounds__(256)
void gcn_gemm_wmma(const _Float16* __restrict__ xh, const _Float16* __restrict__ wh,
                   float* __restrict__ h) {
    const int m0   = blockIdx.x << 4;
    const int lane = threadIdx.x & 31;
    const int n0   = (threadIdx.x >> 5) << 4;   // 0,16,...,112
    const int r    = lane & 15;
    const int hl   = lane >> 4;                 // half-wave selector

    const _Float16* ap = xh + (m0 + r) * GCN_DIM + hl * 8;
    const _Float16* bp = wh + (n0 + r) * GCN_DIM + hl * 8;

    v8f acc = {0.f, 0.f, 0.f, 0.f, 0.f, 0.f, 0.f, 0.f};
#pragma unroll
    for (int kc = 0; kc < GCN_DIM; kc += 32) {
        v8h alo = *(const v8h*)(ap + kc);
        v8h ahi = *(const v8h*)(ap + kc + 16);
        v8h blo = *(const v8h*)(bp + kc);
        v8h bhi = *(const v8h*)(bp + kc + 16);
        v16h A = __builtin_shufflevector(alo, ahi, 0,1,2,3,4,5,6,7,8,9,10,11,12,13,14,15);
        v16h B = __builtin_shufflevector(blo, bhi, 0,1,2,3,4,5,6,7,8,9,10,11,12,13,14,15);
        acc = __builtin_amdgcn_wmma_f32_16x16x32_f16(
            /*neg_a=*/false, A, /*neg_b=*/false, B,
            /*c_mod=*/(short)0, acc, /*reuse_a=*/false, /*reuse_b=*/false);
    }
    // D layout: lane L, vgpr v -> row m0 + v + 8*(L>>4), col n0 + (L&15)
    float* op = h + (m0 + hl * 8) * GCN_DIM + n0 + r;
#pragma unroll
    for (int v = 0; v < 8; ++v) op[v * GCN_DIM] = acc[v];
}

// ---------------- edge scatter: xn[col] += h[row] * dinv[row]*dinv[col] ----------------
// one wave per edge item; items [0,E) are real edges, [E,E+N) are self loops.
__global__ __launch_bounds__(256)
void gcn_scatter(const float* __restrict__ h, const float* __restrict__ dinv,
                 const long long* __restrict__ row, const long long* __restrict__ col,
                 float* __restrict__ xn, int nE, int nN) {
    int item = blockIdx.x * 8 + (threadIdx.x >> 5);
    if (item >= nE + nN) return;
    const int lane = threadIdx.x & 31;
    int r, c; float norm;
    if (item < nE) {
        r = (int)row[item];
        c = (int)col[item];
        __builtin_prefetch(h + (size_t)r * GCN_DIM + lane * 4, 0, 1);
        norm = dinv[r] * dinv[c];
    } else {
        r = c = item - nE;
        float d = dinv[r];
        norm = d * d;
    }
    const float4 hv = *(const float4*)(h + (size_t)r * GCN_DIM + lane * 4);
    float* dst = xn + (size_t)c * GCN_DIM + lane * 4;
    atomicAdd(dst + 0, hv.x * norm);
    atomicAdd(dst + 1, hv.y * norm);
    atomicAdd(dst + 2, hv.z * norm);
    atomicAdd(dst + 3, hv.w * norm);
}

// ---------------- global add pool over graphs ----------------
__global__ __launch_bounds__(256)
void gcn_pool(const float* __restrict__ x, const long long* __restrict__ batch,
              float* __restrict__ out, int n) {
    int node = blockIdx.x * 8 + (threadIdx.x >> 5);
    if (node >= n) return;
    const int lane = threadIdx.x & 31;
    const int g = (int)batch[node];
    const float4 v = *(const float4*)(x + (size_t)node * GCN_DIM + lane * 4);
    float* dst = out + g * GCN_DIM + lane * 4;
    atomicAdd(dst + 0, v.x);
    atomicAdd(dst + 1, v.y);
    atomicAdd(dst + 2, v.z);
    atomicAdd(dst + 3, v.w);
}

// ---------------- host launch ----------------
extern "C" void kernel_launch(void* const* d_in, const int* in_sizes, int n_in,
                              void* d_out, int out_size, void* d_ws, size_t ws_size,
                              hipStream_t stream) {
    (void)in_sizes; (void)n_in; (void)out_size; (void)ws_size;

    const float*     x     = (const float*)d_in[0];                 // [N,128] f32
    const float*     W     = (const float*)d_in[1];                 // [3,128,128] f32
    const float*     b     = (const float*)d_in[2];                 // [3,128] f32
    const long long* edge  = (const long long*)d_in[3];             // [2,E] i64
    const long long* batch = (const long long*)d_in[4];             // [N] i64
    const long long* erow  = edge;                                  // sources
    const long long* ecol  = edge + GCN_N_EDGES;                    // targets

    // workspace layout (all offsets 16B-aligned; total ~64.3 MB)
    char* ws = (char*)d_ws;
    float*     F0   = (float*)(ws);                                 // h buffer   25.6 MB
    float*     F1   = (float*)(ws + 25600000);                      // x ping     25.6 MB
    _Float16*  H16  = (_Float16*)(ws + 51200000);                   // f16 x      12.8 MB
    float*     DINV = (float*)(ws + 64000000);                      // deg/dinv   0.2 MB
    _Float16*  WH   = (_Float16*)(ws + 64200000);                   // f16 W      96 KB

    const int NXE   = GCN_N_NODES * GCN_DIM;        // 6,400,000
    const int NX4   = NXE / 4;                      // 1,600,000
    const int ITEMS = GCN_N_EDGES + GCN_N_NODES;    // 850,000

    // degree -> dinv
    gcn_fill_f32<<<(GCN_N_NODES + 255) / 256, 256, 0, stream>>>(DINV, 0.0f, GCN_N_NODES);
    gcn_deg_accum<<<(GCN_N_EDGES + 255) / 256, 256, 0, stream>>>(ecol, DINV, GCN_N_EDGES);
    gcn_make_dinv<<<(GCN_N_NODES + 255) / 256, 256, 0, stream>>>(DINV, GCN_N_NODES);

    // convert all layer weights to f16 once (3*128*128 / 4 = 12288 threads)
    gcn_cvt_f16<<<48, 256, 0, stream>>>(W, WH, GCN_LAYERS * GCN_DIM * GCN_DIM / 4);

    const float* xcur = x;
    for (int l = 0; l < GCN_LAYERS; ++l) {
        // xcur (f32) -> H16 (f16); after this, the buffer behind xcur is dead
        gcn_cvt_f16<<<(NX4 + 255) / 256, 256, 0, stream>>>(xcur, H16, NX4);
        // h = xh @ W_l^T   (WMMA)
        gcn_gemm_wmma<<<GCN_N_NODES / 16, 256, 0, stream>>>(H16, WH + l * GCN_DIM * GCN_DIM, F0);
        // xnext = bias, then scatter-add normalized messages
        gcn_fill_bias<<<(NXE + 255) / 256, 256, 0, stream>>>(F1, b + l * GCN_DIM, NXE);
        gcn_scatter<<<(ITEMS + 7) / 8, 256, 0, stream>>>(F0, DINV, erow, ecol, F1,
                                                         GCN_N_EDGES, GCN_N_NODES);
        xcur = F1;
    }

    // global add pool
    gcn_fill_f32<<<(GCN_GRAPHS * GCN_DIM + 255) / 256, 256, 0, stream>>>(
        (float*)d_out, 0.0f, GCN_GRAPHS * GCN_DIM);
    gcn_pool<<<(GCN_N_NODES + 7) / 8, 256, 0, stream>>>(F1, batch, (float*)d_out, GCN_N_NODES);
}
